// SimplicialAttentionLayer_7172595384925
// MI455X (gfx1250) — matrix operand order repaired and checked by reading, not compiled, not run on hardware
//
#include <hip/hip_runtime.h>
#include <hip/hip_bf16.h>

// ---------------------------------------------------------------------------
// SimplicialAttentionLayer for MI455X (gfx1250, wave32, WMMA + TDM).
//   out = softmax(softmax(H @ K^T / sqrt(D), axis=1) * mask, axis=1) @ V
// N = M = 4096, D = 1024, fp32 in/out. GEMMs run in bf16 WMMA w/ f32 accum.
// B macro-tiles are staged into LDS by the Tensor Data Mover (one DMA per
// block per k-step, double buffered, TENSORcnt-tracked).
// ---------------------------------------------------------------------------

#define N_DIM 4096
#define M_DIM 4096
#define D_DIM 1024

typedef __attribute__((ext_vector_type(16))) __bf16        v16bf;
typedef __attribute__((ext_vector_type(8)))  float         v8f;
typedef __attribute__((ext_vector_type(4)))  unsigned int  v4u;
typedef __attribute__((ext_vector_type(8)))  int           v8i;
typedef __attribute__((ext_vector_type(4)))  int           v4i;

union BfVec { uint4 u[2]; v16bf v; };

// LDS B-tile: 128 rows (columns of C) x 32 k-halves, padded to 40 halves
// (80 B row stride = 64 B data + 16 B TDM pad) to spread LDS banks.
#define BT_ROWS 128
#define BT_K    32
#define BT_PADK 40

__device__ __forceinline__ unsigned short f2bf(float f) {
  unsigned int u = __float_as_uint(f);
  unsigned int r = u + 0x7FFFu + ((u >> 16) & 1u);   // round-to-nearest-even
  return (unsigned short)(r >> 16);
}

// ------------------------- fp32 -> bf16 convert ----------------------------
__global__ __launch_bounds__(256)
void k_f32_to_bf16(const float* __restrict__ x, unsigned short* __restrict__ y, int n) {
  int i = blockIdx.x * 256 + threadIdx.x;
  if (i < n) y[i] = f2bf(x[i]);
}

// --------------------- V [M,D] f32 -> Vt [D,M] bf16 ------------------------
__global__ __launch_bounds__(256)
void k_transpose_bf16(const float* __restrict__ V, unsigned short* __restrict__ Vt) {
  __shared__ float tile[32][33];
  const int d0 = blockIdx.x * 32;
  const int m0 = blockIdx.y * 32;
  const int tx = threadIdx.x;      // 0..31
  const int ty = threadIdx.y;      // 0..7
#pragma unroll
  for (int i = 0; i < 32; i += 8)
    tile[ty + i][tx] = V[(size_t)(m0 + ty + i) * D_DIM + d0 + tx];
  __syncthreads();
#pragma unroll
  for (int i = 0; i < 32; i += 8)
    Vt[(size_t)(d0 + ty + i) * M_DIM + m0 + tx] = f2bf(tile[tx][ty + i]);
}

// ------------------- TDM: stage B tile (BT_ROWS x BT_K) --------------------
// D# per CDNA5 ISA ch.8: group0 = {count, lds_addr, global_addr, type=2},
// group1 = {data_size=2B, pad(4 DW every 16 DW), tensor dims, tile dims,
// dim0 stride}, groups 2/3 = 0 (2-D tensor).
// amdgpu-toolchain (clang-23) 6-arg builtin:
//   (uint32x4 g0, int32x8 g1, int32x4 g2, int32x4 g3, int32x8, i32 cpol)
__device__ __forceinline__ void tdm_load_b_tile(unsigned lds_addr,
                                                const unsigned short* gptr,
                                                int ldb) {
  const unsigned long long ga = (unsigned long long)(uintptr_t)gptr;
  v4u g0 = {
    1u,                                                   // count=1, no gather
    lds_addr,                                             // LDS byte address
    (unsigned)(ga & 0xFFFFFFFFull),                       // global_addr[31:0]
    (unsigned)((ga >> 32) & 0x01FFFFFFull) | (2u << 30)   // [56:32] | type=2
  };
  const unsigned td0 = (unsigned)ldb;       // tensor_dim0 (elements)
  const unsigned td1 = 0x40000000u;         // tensor_dim1: huge, no OOB clip
  const unsigned long long st0 = (unsigned long long)(unsigned)ldb; // stride0
  v8i g1 = {
    (int)0x06D10000u,                                     // data_size=1(2B), pad_en, int=3(16DW), amt=3(4DW)
    (int)((td0 & 0xFFFFu) << 16),                         // abar=0 | dim0.lo
    (int)((td0 >> 16) | ((td1 & 0xFFFFu) << 16)),         // dim0.hi | dim1.lo
    (int)((td1 >> 16) | ((unsigned)BT_K << 16)),          // dim1.hi | tile_dim0
    (int)BT_ROWS,                                         // tile_dim1 | tile_dim2=0
    (int)(st0 & 0xFFFFFFFFull),                           // stride0[31:0]
    (int)((st0 >> 32) & 0xFFFFull),                       // stride0[47:32] | stride1.lo=0
    0                                                     // stride1.hi
  };
  v4i z4 = {0, 0, 0, 0};
  v8i z8 = {0, 0, 0, 0, 0, 0, 0, 0};
  __builtin_amdgcn_tensor_load_to_lds(g0, g1, z4, z4, z8, 0);
}

// ----------------------- WMMA fragment loaders -----------------------------
// A fragment (16x32 bf16, ISA 7.12.2): lane holds row lane%16;
// lanes 0-15: K = {k0..k0+7, k0+16..k0+23}; lanes 16-31: +8.
__device__ __forceinline__ v16bf load_a_frag(const unsigned short* __restrict__ base,
                                             int lda, int row0, int k0, int lane) {
  const int r  = row0 + (lane & 15);
  const int kb = k0 + ((lane >> 4) << 3);
  const unsigned short* p = base + (size_t)r * lda + kb;
  BfVec x;
  x.u[0] = *reinterpret_cast<const uint4*>(p);
  x.u[1] = *reinterpret_cast<const uint4*>(p + 16);
  return x.v;
}

// B fragment (32x16 bf16) from the LDS-staged tile: lane holds column
// lane%16 (tile row), K = {kb..kb+7, kb+16..kb+23} (kb=8 for upper lanes).
__device__ __forceinline__ v16bf load_b_lds(const unsigned short* bs,
                                            int crel, int lane) {
  const int c  = crel + (lane & 15);
  const int kb = (lane >> 4) << 3;
  const unsigned short* p = bs + c * BT_PADK + kb;
  BfVec x;
  x.u[0] = *reinterpret_cast<const uint4*>(p);
  x.u[1] = *reinterpret_cast<const uint4*>(p + 16);
  return x.v;
}

// C/D (16x16 f32): lane L holds column L%16, rows v + (L>=16 ? 8 : 0).
__device__ __forceinline__ void store_tile(float* __restrict__ C, int ldc,
                                           const v8f& acc, int r0, int c, float scale) {
#pragma unroll
  for (int v = 0; v < 8; ++v)
    C[(size_t)(r0 + v) * ldc + c] = acc[v] * scale;
}

// ------------------- C[r,c] = scale * sum_k A[r,k]*B[c,k] ------------------
// Both operands row-major with contiguous K. 8 waves/block; each wave does a
// 32x32 output tile (2x2 WMMA tiles), block tile = 64 rows x 128 cols.
// Wave 0 drives the TDM double-buffered staging of the 128x32 B tile.
__global__ __launch_bounds__(256)
void k_wmma_gemm_abT(const unsigned short* __restrict__ A,
                     const unsigned short* __restrict__ B,
                     float* __restrict__ C,
                     int Kdim, int lda, int ldb, int ldc, float scale) {
  __shared__ unsigned short Bs[2][BT_ROWS * BT_PADK];

  const int lane = threadIdx.x & 31;
  const int wave = threadIdx.x >> 5;
  const bool lead = (threadIdx.x < 32);          // wave 0 issues TDM

  const int row0   = blockIdx.y * 64 + (wave >> 2) * 32;
  const int colblk = blockIdx.x * 128;           // block's B-tile base column
  const int crel   = (wave & 3) * 32;            // wave's column offset in tile

  const unsigned short* __restrict__ Bsrc = B + (size_t)colblk * ldb;

  v8f acc00 = {}, acc01 = {}, acc10 = {}, acc11 = {};

  if (lead) {
    tdm_load_b_tile((unsigned)(uintptr_t)&Bs[0][0], Bsrc, ldb);
    __builtin_amdgcn_s_wait_tensorcnt(0);
  }
  __syncthreads();

  int cur = 0;
  for (int k0 = 0; k0 < Kdim; k0 += BT_K) {
    const int  nxt      = cur ^ 1;
    const bool has_next = (k0 + BT_K) < Kdim;
    if (lead && has_next)                         // prefetch next B tile (DMA
      tdm_load_b_tile((unsigned)(uintptr_t)&Bs[nxt][0],   // overlaps compute)
                      Bsrc + (k0 + BT_K), ldb);

    v16bf a0 = load_a_frag(A, lda, row0,      k0, lane);
    v16bf a1 = load_a_frag(A, lda, row0 + 16, k0, lane);
    v16bf b0 = load_b_lds(&Bs[cur][0], crel,      lane);
    v16bf b1 = load_b_lds(&Bs[cur][0], crel + 16, lane);
    acc00 = __builtin_amdgcn_wmma_f32_16x16x32_bf16(false, a0, false, b0, (short)0, acc00, false, false);
    acc01 = __builtin_amdgcn_wmma_f32_16x16x32_bf16(false, a0, false, b1, (short)0, acc01, false, false);
    acc10 = __builtin_amdgcn_wmma_f32_16x16x32_bf16(false, a1, false, b0, (short)0, acc10, false, false);
    acc11 = __builtin_amdgcn_wmma_f32_16x16x32_bf16(false, a1, false, b1, (short)0, acc11, false, false);

    if (lead && has_next) __builtin_amdgcn_s_wait_tensorcnt(0);
    __syncthreads();
    cur = nxt;
  }

  const int cc = lane & 15;
  const int rs = (lane >> 4) << 3;
  const int col0 = colblk + crel;
  store_tile(C, ldc, acc00, row0 + rs,      col0 + cc,      scale);
  store_tile(C, ldc, acc01, row0 + rs,      col0 + 16 + cc, scale);
  store_tile(C, ldc, acc10, row0 + 16 + rs, col0 + cc,      scale);
  store_tile(C, ldc, acc11, row0 + 16 + rs, col0 + 16 + cc, scale);
}

// --------------------------- double softmax --------------------------------
__device__ __forceinline__ float wave_max(float v) {
#pragma unroll
  for (int o = 16; o > 0; o >>= 1) v = fmaxf(v, __shfl_xor(v, o, 32));
  return v;
}
__device__ __forceinline__ float wave_sum(float v) {
#pragma unroll
  for (int o = 16; o > 0; o >>= 1) v += __shfl_xor(v, o, 32);
  return v;
}
__device__ float block_reduce(float v, bool is_max) {
  __shared__ float red[8];
  v = is_max ? wave_max(v) : wave_sum(v);
  const int wave = threadIdx.x >> 5, lane = threadIdx.x & 31;
  if (lane == 0) red[wave] = v;
  __syncthreads();
  float r = red[0];
#pragma unroll
  for (int w = 1; w < 8; ++w) r = is_max ? fmaxf(r, red[w]) : (r + red[w]);
  __syncthreads();   // allow shared reuse on the next call
  return r;
}

// One block (256 threads, 8 waves) per row; row held in registers.
__global__ __launch_bounds__(256)
void k_double_softmax(const float* __restrict__ S, const float* __restrict__ mask,
                      unsigned short* __restrict__ P) {
  const int n = blockIdx.x;
  const int t = threadIdx.x;
  const int PER = M_DIM / 256;                 // 16 elements / thread
  const float* __restrict__ srow = S    + (size_t)n * M_DIM;
  const float* __restrict__ mrow = mask + (size_t)n * M_DIM;

  float s[PER];
  float mx = -3.402823466e38f;
#pragma unroll
  for (int i = 0; i < PER; ++i) { s[i] = srow[t + i * 256]; mx = fmaxf(mx, s[i]); }
  mx = block_reduce(mx, true);

  float sum = 0.f;
#pragma unroll
  for (int i = 0; i < PER; ++i) { s[i] = __expf(s[i] - mx); sum += s[i]; }
  sum = block_reduce(sum, false);
  const float inv = 1.f / sum;

  float mx2 = -3.402823466e38f;
#pragma unroll
  for (int i = 0; i < PER; ++i) {
    s[i] = s[i] * inv * mrow[t + i * 256];
    mx2 = fmaxf(mx2, s[i]);
  }
  mx2 = block_reduce(mx2, true);

  float sum2 = 0.f;
#pragma unroll
  for (int i = 0; i < PER; ++i) { s[i] = __expf(s[i] - mx2); sum2 += s[i]; }
  sum2 = block_reduce(sum2, false);
  const float inv2 = 1.f / sum2;

  unsigned short* __restrict__ prow = P + (size_t)n * M_DIM;
#pragma unroll
  for (int i = 0; i < PER; ++i) prow[t + i * 256] = f2bf(s[i] * inv2);
}

// ------------------------------- launcher ----------------------------------
extern "C" void kernel_launch(void* const* d_in, const int* in_sizes, int n_in,
                              void* d_out, int out_size, void* d_ws, size_t ws_size,
                              hipStream_t stream) {
  const float* H    = (const float*)d_in[0];   // [N, D]
  const float* K    = (const float*)d_in[1];   // [M, D]
  const float* V    = (const float*)d_in[2];   // [M, D]
  const float* mask = (const float*)d_in[3];   // [N, M]
  float* out = (float*)d_out;                  // [N, D]

  // Workspace layout (120 MB total):
  char* ws = (char*)d_ws;
  unsigned short* Hb = (unsigned short*)(ws);                          //  8 MB bf16 [N,D]
  unsigned short* Kb = (unsigned short*)(ws + ((size_t) 8 << 20));     //  8 MB bf16 [M,D]
  unsigned short* Vt = (unsigned short*)(ws + ((size_t)16 << 20));     //  8 MB bf16 [D,M]
  float*          S  = (float*)         (ws + ((size_t)24 << 20));     // 64 MB f32  [N,M]
  unsigned short* P  = (unsigned short*)(ws + ((size_t)88 << 20));     // 32 MB bf16 [N,M]

  // 1) bf16 conversions
  k_f32_to_bf16<<<(N_DIM * D_DIM) / 256, 256, 0, stream>>>(H, Hb, N_DIM * D_DIM);
  k_f32_to_bf16<<<(M_DIM * D_DIM) / 256, 256, 0, stream>>>(K, Kb, M_DIM * D_DIM);
  k_transpose_bf16<<<dim3(D_DIM / 32, M_DIM / 32), dim3(32, 8), 0, stream>>>(V, Vt);

  // 2) S = Hb @ Kb^T / sqrt(D)
  k_wmma_gemm_abT<<<dim3(M_DIM / 128, N_DIM / 64), 256, 0, stream>>>(
      Hb, Kb, S, D_DIM, D_DIM, D_DIM, M_DIM, 0.03125f /* 1/sqrt(1024) */);

  // 3) P = softmax(softmax(S) * mask)   (bf16 output)
  k_double_softmax<<<N_DIM, 256, 0, stream>>>(S, mask, P);

  // 4) out = P @ V  (via Vt so both operands are K-contiguous)
  k_wmma_gemm_abT<<<dim3(D_DIM / 128, N_DIM / 64), 256, 0, stream>>>(
      P, Vt, out, M_DIM, M_DIM, M_DIM, D_DIM, 1.0f);
}
